// SelfAttention_82935818485797
// MI455X (gfx1250) — matrix operand order repaired and checked by reading, not compiled, not run on hardware
//
#include <hip/hip_runtime.h>
#include <hip/hip_bf16.h>
#include <hip/hip_fp16.h>

// ---------------------------------------------------------------------------
// Self-attention (B=4, C=256, H=W=64 -> N=4096, Cq=32) for gfx1250 (CDNA5).
// Flash-attention with v_wmma_f32_16x16x32_f16, online softmax, and
// double-buffered GLOBAL_LOAD_ASYNC_TO_LDS staging of K/V blocks shared by
// the 4 waves of each workgroup.
// ---------------------------------------------------------------------------

typedef __attribute__((ext_vector_type(16))) _Float16 v16h;
typedef __attribute__((ext_vector_type(8)))  _Float16 v8h;
typedef __attribute__((ext_vector_type(8)))  float    v8f;
typedef __attribute__((ext_vector_type(4)))  int      v4i;

// Address-space-qualified int4 for the async-copy builtin's typed pointers.
typedef __attribute__((address_space(1))) v4i gas_v4i;   // global
typedef __attribute__((address_space(3))) v4i las_v4i;   // LDS

#define B_   4
#define C_   256
#define N_   4096   // H*W
#define CQ_  32
#define LOG2N_ 12   // N_ = 4096
#define LDS_PITCH 40   // f16 row pitch: 80B = 20 banks -> conflict-free halves

static_assert(C_ % 32 == 0 && N_ % 32 == 0 && CQ_ == 32, "tiling assumptions");

// ---- CDNA5 async global->LDS path (guarded; falls back to sync staging) ---
#if defined(__has_builtin)
#  if __has_builtin(__builtin_amdgcn_global_load_async_to_lds_b128) && \
      __has_builtin(__builtin_amdgcn_s_wait_asynccnt)
#    define HAS_ASYNC 1
#  endif
#endif
#ifndef HAS_ASYNC
#  define HAS_ASYNC 0
#endif

__device__ __forceinline__ void cp_async_b128(_Float16* lds_dst,
                                              const _Float16* gsrc) {
#if HAS_ASYNC
    __builtin_amdgcn_global_load_async_to_lds_b128(
        (gas_v4i*)gsrc, (las_v4i*)lds_dst, /*imm offset*/0, /*cpol*/0);
#else
    *(v8h*)lds_dst = *(const v8h*)gsrc;
#endif
}

__device__ __forceinline__ void wait_async_all() {
#if HAS_ASYNC
    __builtin_amdgcn_s_wait_asynccnt(0);
#endif
}

__device__ __forceinline__ v8f wmma_f16(v16h a, v16h b, v8f c) {
    // (neg_a, A, neg_b, B, c_mod, C, reuse_a, reuse_b)
    return __builtin_amdgcn_wmma_f32_16x16x32_f16(false, a, false, b, (short)0, c,
                                                  false, false);
}

// A fragment: 16x32 f16 tile, row-major source with leading dim `ld`.
// Lane L holds row M = L%16; halves [0..7] = K kb..kb+7, [8..15] = K 16+kb..,
// kb = (L<16 ? 0 : 8).   (ISA 7.12.2, 16-bit A-matrix 16x32)
__device__ __forceinline__ v16h load_a_frag(const _Float16* __restrict__ base,
                                            int ld, int lane) {
    const int m  = lane & 15;
    const int kb = (lane < 16) ? 0 : 8;
    const _Float16* row = base + (size_t)m * ld + kb;
    union { v16h v; v8h h2[2]; } f;
    f.h2[0] = *(const v8h*)(row);
    f.h2[1] = *(const v8h*)(row + 16);
    return f.v;
}

// B fragment: 32x16 (KxN). Lane L holds column N = L%16; halves [0..7] = K kb..kb+7,
// [8..15] = K 16+kb.., kb = (L<16 ? 0 : 8). `base` points at a row-major
// [16 x >=32] array whose row n is column n of B (i.e. B^T rows).
__device__ __forceinline__ v16h load_b_frag(const _Float16* __restrict__ base,
                                            int ld, int lane) {
    const int c  = lane & 15;
    const int kb = (lane < 16) ? 0 : 8;
    const _Float16* row = base + (size_t)c * ld + kb;
    union { v16h v; v8h h2[2]; } f;
    f.h2[0] = *(const v8h*)(row);
    f.h2[1] = *(const v8h*)(row + 16);
    return f.v;
}

// Same, but from an LDS tile with LDS_PITCH row pitch (rowbase = &tile[col0][0]).
__device__ __forceinline__ v16h load_b_frag_lds(const _Float16* rowbase, int lane) {
    const int c  = lane & 15;
    const int kb = (lane < 16) ? 0 : 8;
    const _Float16* p = rowbase + (size_t)c * LDS_PITCH + kb;
    union { v16h v; v8h h2[2]; } f;
    f.h2[0] = *(const v8h*)(p);
    f.h2[1] = *(const v8h*)(p + 16);
    return f.v;
}

// ---------------------------------------------------------------------------
// Kernel 0a: f32 -> f16 elementwise convert (weight matrices).
// ---------------------------------------------------------------------------
__global__ void convert_f32_f16(const float* __restrict__ src,
                                _Float16* __restrict__ dst, int n) {
    int i = blockIdx.x * blockDim.x + threadIdx.x;
    if (i < n) dst[i] = (_Float16)src[i];
}

// ---------------------------------------------------------------------------
// Kernel 0b: x [B,C,N] f32 -> xt [B,N,C] f16   (LDS-tiled transpose, 32x32)
// ---------------------------------------------------------------------------
__global__ __launch_bounds__(128) void transpose_convert_x(
        const float* __restrict__ x, _Float16* __restrict__ xt) {
    __shared__ _Float16 tile[32][33];
    const int b  = blockIdx.z;
    const int c0 = blockIdx.y * 32;
    const int n0 = blockIdx.x * 32;
    const int tx = threadIdx.x & 31;
    const int ty = threadIdx.x >> 5;          // 0..3

    const float* src = x + ((size_t)b * C_ + c0) * N_ + n0;
    for (int i = ty; i < 32; i += 4)
        tile[i][tx] = (_Float16)src[(size_t)i * N_ + tx];
    __syncthreads();
    _Float16* dst = xt + ((size_t)b * N_ + n0) * C_ + c0;
    for (int i = ty; i < 32; i += 4)
        dst[(size_t)i * C_ + tx] = tile[tx][i];
}

// ---------------------------------------------------------------------------
// Kernel 1: fused q/k/v projection.
//   qh [B*N, 32]  row-major f16   (q[b,n,:])
//   kh [B*N, 32]  row-major f16   (k[b,:,n] transposed -> row n)
//   vT [B, C, N]  f16             (v transposed: vT[b][cv][n])
// One wave computes 16 pixel rows: 20 output tiles x 8 K-steps = 160 WMMAs.
// ---------------------------------------------------------------------------
__global__ __launch_bounds__(128) void proj_qkv(
        const _Float16* __restrict__ xt,
        const _Float16* __restrict__ Whq, const float* __restrict__ bq,
        const _Float16* __restrict__ Whk, const float* __restrict__ bk,
        const _Float16* __restrict__ Whv, const float* __restrict__ bv,
        _Float16* __restrict__ qh, _Float16* __restrict__ kh,
        _Float16* __restrict__ vT) {
    const int wave = (blockIdx.x * blockDim.x + threadIdx.x) >> 5; // 0..B*N/16-1
    const int lane = threadIdx.x & 31;
    const int row0 = wave * 16;                 // global pixel row = b*N + n
    const int cl   = lane & 15;
    const int rofs = (lane < 16) ? 0 : 8;
    const int b    = row0 >> LOG2N_;
    const int n0   = row0 & (N_ - 1);

    // Keep all 8 A-fragments (K = 256) resident.
    v16h a[8];
    #pragma unroll
    for (int kk = 0; kk < 8; ++kk)
        a[kk] = load_a_frag(xt + (size_t)row0 * C_ + kk * 32, C_, lane);

    // ---- q and k: 2 column tiles each -------------------------------------
    #pragma unroll
    for (int t = 0; t < 2; ++t) {
        v8f accq = {}; v8f acck = {};
        #pragma unroll
        for (int kk = 0; kk < 8; ++kk) {
            accq = wmma_f16(a[kk], load_b_frag(Whq + (size_t)(t*16)*C_ + kk*32, C_, lane), accq);
            acck = wmma_f16(a[kk], load_b_frag(Whk + (size_t)(t*16)*C_ + kk*32, C_, lane), acck);
        }
        const float biasq = bq[t * 16 + cl];
        const float biask = bk[t * 16 + cl];
        #pragma unroll
        for (int r = 0; r < 8; ++r) {
            const size_t row = (size_t)(row0 + r + rofs);
            qh[row * CQ_ + t * 16 + cl] = (_Float16)(accq[r] + biasq);
            kh[row * CQ_ + t * 16 + cl] = (_Float16)(acck[r] + biask);
        }
    }

    // ---- v: 16 column tiles, stored transposed ----------------------------
    #pragma unroll 4
    for (int t = 0; t < 16; ++t) {
        v8f acc = {};
        #pragma unroll
        for (int kk = 0; kk < 8; ++kk)
            acc = wmma_f16(a[kk], load_b_frag(Whv + (size_t)(t*16)*C_ + kk*32, C_, lane), acc);
        const float bias = bv[t * 16 + cl];
        union { _Float16 h[8]; v8h v; } tmp;
        #pragma unroll
        for (int r = 0; r < 8; ++r) tmp.h[r] = (_Float16)(acc[r] + bias);
        // lane owns column cv = t*16+cl, rows n0+rofs .. n0+rofs+7 contiguous
        *(v8h*)(vT + ((size_t)b * C_ + t * 16 + cl) * N_ + n0 + rofs) = tmp.v;
    }
}

// ---------------------------------------------------------------------------
// Stage one 32-key block of K (32x32 f16) and V (256x32 f16) into LDS.
// 128 threads, 9 async b128 per thread (1 for K, 8 for V).
// ---------------------------------------------------------------------------
__device__ __forceinline__ void stage_kv(_Float16 (*kT)[LDS_PITCH],
                                         _Float16 (*vTl)[LDS_PITCH],
                                         const _Float16* kbase,
                                         const _Float16* vbase,
                                         int j, int tid) {
    {   // K: 32 rows x 64B = 128 x 16B units
        const int row = tid >> 2, seg = tid & 3;
        cp_async_b128(&kT[row][seg * 8], kbase + (size_t)(j + row) * CQ_ + seg * 8);
    }
    #pragma unroll
    for (int s = 0; s < 8; ++s) {   // V: 256 rows x 64B = 1024 x 16B units
        const int u = tid + s * 128;
        const int cv = u >> 2, seg = u & 3;
        cp_async_b128(&vTl[cv][seg * 8], vbase + (size_t)cv * N_ + j + seg * 8);
    }
}

// ---------------------------------------------------------------------------
// Kernel 2: flash attention. 4 waves/block share K/V LDS tiles; each wave
// owns 16 query rows x full 256 V dim. Per 32-key block: 2 WMMAs (S) + 16
// WMMAs (P@V), with the next block's K/V copy running asynchronously.
// ---------------------------------------------------------------------------
#define ATT_WAVES 4
__global__ __launch_bounds__(32 * ATT_WAVES) void attention(
        const _Float16* __restrict__ qh, const _Float16* __restrict__ kh,
        const _Float16* __restrict__ vT, float* __restrict__ out) {
    __shared__ _Float16 kTile[2][32][LDS_PITCH];     //  5 KB
    __shared__ _Float16 vTile[2][C_][LDS_PITCH];     // 40 KB
    __shared__ _Float16 plds[ATT_WAVES][16][LDS_PITCH]; // 5 KB (P re-layout)

    const int tid  = threadIdx.x;
    const int wib  = tid >> 5;
    const int lane = tid & 31;
    const int wave = blockIdx.x * ATT_WAVES + wib;
    const int row0 = wave * 16;                    // global row = b*N + n0
    const int b    = row0 >> LOG2N_;               // same b for all 4 waves
    const int cl   = lane & 15;
    const int rofs = (lane < 16) ? 0 : 8;

    const _Float16* kbase = kh + (size_t)b * N_ * CQ_;
    const _Float16* vbase = vT + (size_t)b * C_ * N_;

    const v16h qa = load_a_frag(qh + (size_t)row0 * CQ_, CQ_, lane);

    v8f acc[16];
    #pragma unroll
    for (int t = 0; t < 16; ++t) { v8f z = {}; acc[t] = z; }
    float mrun[8], lrun[8];
    #pragma unroll
    for (int r = 0; r < 8; ++r) { mrun[r] = -1e30f; lrun[r] = 0.0f; }

    const float sc = 0.015625f;                    // 1/sqrt(4096)
    const int NBLK = N_ / 32;

    stage_kv(kTile[0], vTile[0], kbase, vbase, 0, tid);   // prologue

    for (int jb = 0; jb < NBLK; ++jb) {
        const int buf = jb & 1;
        wait_async_all();      // this wave's copies of block jb are in LDS
        __syncthreads();       // all waves' copies done; buf^1 no longer read
        if (jb + 1 < NBLK)     // overlap next copy with this block's compute
            stage_kv(kTile[buf ^ 1], vTile[buf ^ 1], kbase, vbase,
                     (jb + 1) * 32, tid);

        // ---- S = (Q K^T)/sqrt(N): two 16x16 f32 tiles from LDS ------------
        v16h kb0 = load_b_frag_lds(&kTile[buf][0][0],  lane);
        v16h kb1 = load_b_frag_lds(&kTile[buf][16][0], lane);
        v8f z = {};
        v8f s0 = wmma_f16(qa, kb0, z);
        v8f s1 = wmma_f16(qa, kb1, z);

        float rmax[8];
        #pragma unroll
        for (int r = 0; r < 8; ++r) {
            s0[r] *= sc; s1[r] *= sc;
            rmax[r] = fmaxf(s0[r], s1[r]);
        }
        #pragma unroll
        for (int off = 1; off < 16; off <<= 1)
            #pragma unroll
            for (int r = 0; r < 8; ++r)
                rmax[r] = fmaxf(rmax[r], __shfl_xor(rmax[r], off, 32));

        // ---- online softmax ----------------------------------------------
        float alpha[8], rsum[8];
        #pragma unroll
        for (int r = 0; r < 8; ++r) {
            const float mnew = fmaxf(mrun[r], rmax[r]);
            alpha[r] = __expf(mrun[r] - mnew);
            mrun[r]  = mnew;
            s0[r] = __expf(s0[r] - mnew);
            s1[r] = __expf(s1[r] - mnew);
            rsum[r] = s0[r] + s1[r];
        }
        #pragma unroll
        for (int off = 1; off < 16; off <<= 1)
            #pragma unroll
            for (int r = 0; r < 8; ++r)
                rsum[r] += __shfl_xor(rsum[r], off, 32);
        #pragma unroll
        for (int r = 0; r < 8; ++r) lrun[r] = lrun[r] * alpha[r] + rsum[r];
        #pragma unroll
        for (int t = 0; t < 16; ++t)
            #pragma unroll
            for (int r = 0; r < 8; ++r) acc[t][r] *= alpha[r];

        // ---- re-layout P (D-frag f32) -> A-frag f16 via wave-private LDS --
        _Float16 (*P)[LDS_PITCH] = plds[wib];
        #pragma unroll
        for (int r = 0; r < 8; ++r) {
            P[r + rofs][cl]      = (_Float16)s0[r];
            P[r + rofs][16 + cl] = (_Float16)s1[r];
        }
        union { v16h v; _Float16 h[16]; } pa;
        {
            const int kb2 = (lane < 16) ? 0 : 8;
            #pragma unroll
            for (int i = 0; i < 8; ++i) {
                pa.h[i]     = P[cl][kb2 + i];
                pa.h[8 + i] = P[cl][16 + kb2 + i];
            }
        }

        // ---- acc += P @ V : 16 column tiles from LDS ----------------------
        #pragma unroll 4
        for (int t = 0; t < 16; ++t) {
            v16h vb = load_b_frag_lds(&vTile[buf][t * 16][0], lane);
            acc[t] = wmma_f16(pa.v, vb, acc[t]);
        }
    }

    // ---- normalize + store: out flat layout is [B, N, C] ------------------
    float* obase = out + (size_t)row0 * C_;
    #pragma unroll
    for (int r = 0; r < 8; ++r) {
        const float inv = 1.0f / lrun[r];
        #pragma unroll
        for (int t = 0; t < 16; ++t)
            obase[(size_t)(r + rofs) * C_ + t * 16 + cl] = acc[t][r] * inv;
    }
}

// ---------------------------------------------------------------------------
extern "C" void kernel_launch(void* const* d_in, const int* in_sizes, int n_in,
                              void* d_out, int out_size, void* d_ws, size_t ws_size,
                              hipStream_t stream) {
    const float* x  = (const float*)d_in[0];
    const float* Wq = (const float*)d_in[1];
    const float* bq = (const float*)d_in[2];
    const float* Wk = (const float*)d_in[3];
    const float* bk = (const float*)d_in[4];
    const float* Wv = (const float*)d_in[5];
    const float* bv = (const float*)d_in[6];
    float* out = (float*)d_out;

    // Workspace carve-up (f16), 256 B aligned; total ~18.2 MB.
    char* ws = (char*)d_ws;
    size_t off = 0;
    auto carve = [&](size_t bytes) {
        off = (off + 255) & ~(size_t)255;
        void* p = ws + off;
        off += bytes;
        return p;
    };
    _Float16* xt  = (_Float16*)carve((size_t)B_ * N_ * C_  * 2);
    _Float16* qh  = (_Float16*)carve((size_t)B_ * N_ * CQ_ * 2);
    _Float16* kh  = (_Float16*)carve((size_t)B_ * N_ * CQ_ * 2);
    _Float16* vT  = (_Float16*)carve((size_t)B_ * C_ * N_  * 2);
    _Float16* Whq = (_Float16*)carve((size_t)CQ_ * C_ * 2);
    _Float16* Whk = (_Float16*)carve((size_t)CQ_ * C_ * 2);
    _Float16* Whv = (_Float16*)carve((size_t)C_  * C_ * 2);
    (void)ws_size; (void)in_sizes; (void)n_in; (void)out_size;

    // 1) weights -> f16
    convert_f32_f16<<<(CQ_ * C_ + 255) / 256, 256, 0, stream>>>(Wq, Whq, CQ_ * C_);
    convert_f32_f16<<<(CQ_ * C_ + 255) / 256, 256, 0, stream>>>(Wk, Whk, CQ_ * C_);
    convert_f32_f16<<<(C_  * C_ + 255) / 256, 256, 0, stream>>>(Wv, Whv, C_ * C_);

    // 2) x [B,C,N] -> xt [B,N,C] f16
    dim3 tg(N_ / 32, C_ / 32, B_);
    transpose_convert_x<<<tg, 128, 0, stream>>>(x, xt);

    // 3) fused q/k/v projection: B*N/16 waves, 4 waves/block
    proj_qkv<<<(B_ * N_ / 16) / 4, 128, 0, stream>>>(xt, Whq, bq, Whk, bk, Whv, bv,
                                                     qh, kh, vT);

    // 4) flash attention: B*N/16 waves, 4 waves/block
    attention<<<(B_ * N_ / 16) / ATT_WAVES, 32 * ATT_WAVES, 0, stream>>>(qh, kh, vT, out);
}
